// _GPTAttn_41180146434219
// MI455X (gfx1250) — matrix operand order repaired and checked by reading, not compiled, not run on hardware
//
#include <hip/hip_runtime.h>

// ---------- types ----------
typedef __bf16 bf16_t;
typedef __attribute__((ext_vector_type(16))) __bf16 v16bf;
typedef __attribute__((ext_vector_type(8)))  __bf16 v8bf;
typedef __attribute__((ext_vector_type(4)))  __bf16 v4bf;
typedef __attribute__((ext_vector_type(8)))  float  v8f;
typedef __attribute__((ext_vector_type(4)))  unsigned int u32x4;

#define B_  4
#define L_  2048
#define D_  1024
#define H_  16
#define HD_ 64
#define BH_ (B_ * H_)
#define M_ROWS (B_ * L_)             // 8192
// fold 1/sqrt(HD) and log2(e) into Q so softmax uses exp2
#define QSCALE 0.1803368801111226f  // 1.4426950408889634 / 8

__device__ __forceinline__ v16bf frag_cat(v8bf lo, v8bf hi) {
  return __builtin_shufflevector(lo, hi, 0,1,2,3,4,5,6,7,8,9,10,11,12,13,14,15);
}
// A operand 16x32: lane holds row m=lane%16; elems 0..7 -> k=half*8+e, 8..15 -> k=16+half*8+(e-8)
__device__ __forceinline__ v16bf load_fragA(const bf16_t* row, int half) {
  v8bf lo = *reinterpret_cast<const v8bf*>(row + half * 8);
  v8bf hi = *reinterpret_cast<const v8bf*>(row + 16 + half * 8);
  return frag_cat(lo, hi);
}
// B operand 32x16: lane holds col n=lane%16; elems 0..7 -> k=half*8+e, 8..15 -> k=16+half*8+(e-8)
__device__ __forceinline__ v16bf load_fragB(const bf16_t* row, int half) {
  v8bf lo = *reinterpret_cast<const v8bf*>(row + half * 8);
  v8bf hi = *reinterpret_cast<const v8bf*>(row + 16 + half * 8);
  return frag_cat(lo, hi);
}
__device__ __forceinline__ v8f wmma_bf16(v16bf a, v16bf b, v8f c) {
  return __builtin_amdgcn_wmma_f32_16x16x32_bf16(false, a, false, b, (short)0, c, false, false);
}

// ---------- stage 0: fp32 -> bf16 ----------
__global__ __launch_bounds__(256) void cvt_kernel(const float* __restrict__ s,
                                                  bf16_t* __restrict__ d, int n4) {
  int i = blockIdx.x * 256 + threadIdx.x;
  if (i < n4) {
    float4 v = reinterpret_cast<const float4*>(s)[i];
    v4bf o = { (bf16_t)v.x, (bf16_t)v.y, (bf16_t)v.z, (bf16_t)v.w };
    *reinterpret_cast<v4bf*>(d + (size_t)i * 4) = o;
  }
}

// ---------- WMMA GEMM: C[M,N] = A[M,K] * Bm[N,K]^T, software-pipelined ----------
// MODE 0: epilogue scatters Q (scaled) / K / V^T (bf16); MODE 1: plain fp32 store
#define LSTR 40   // padded LDS row stride (bf16 elems); 80 B keeps 16-B alignment
template <int MODE>
__global__ __launch_bounds__(256) void gemm_kernel(
    const bf16_t* __restrict__ A, const bf16_t* __restrict__ Bm,
    float* __restrict__ outF,
    bf16_t* __restrict__ q_out, bf16_t* __restrict__ k_out, bf16_t* __restrict__ vt_out,
    int M, int N, int K) {
  __shared__ bf16_t lA[128 * LSTR];
  __shared__ bf16_t lB[128 * LSTR];

  const int tid  = threadIdx.x;
  const int lane = tid & 31;
  const int wave = tid >> 5;
  const int wm   = wave >> 2;   // 0..1
  const int wn   = wave & 3;    // 0..3
  const int m0   = blockIdx.y * 128;
  const int n0   = blockIdx.x * 128;
  const int half = lane >> 4;
  const int l16  = lane & 15;

  // cooperative tile loaders: 128 rows x 4 chunks of 16 B each for A and B
  const int row0 = tid >> 2, ch = tid & 3;
  const bf16_t* aPtr0 = A  + (size_t)(m0 + row0)      * K + ch * 8;
  const bf16_t* aPtr1 = A  + (size_t)(m0 + row0 + 64) * K + ch * 8;
  const bf16_t* bPtr0 = Bm + (size_t)(n0 + row0)      * K + ch * 8;
  const bf16_t* bPtr1 = Bm + (size_t)(n0 + row0 + 64) * K + ch * 8;
  bf16_t* sA0 = &lA[row0 * LSTR + ch * 8];
  bf16_t* sA1 = &lA[(row0 + 64) * LSTR + ch * 8];
  bf16_t* sB0 = &lB[row0 * LSTR + ch * 8];
  bf16_t* sB1 = &lB[(row0 + 64) * LSTR + ch * 8];

  v8f acc[4][2];
#pragma unroll
  for (int mt = 0; mt < 4; ++mt)
#pragma unroll
    for (int nt = 0; nt < 2; ++nt) acc[mt][nt] = (v8f){0,0,0,0,0,0,0,0};

  // prologue: fetch k-block 0 into registers
  u32x4 pa0 = *reinterpret_cast<const u32x4*>(aPtr0);
  u32x4 pa1 = *reinterpret_cast<const u32x4*>(aPtr1);
  u32x4 pb0 = *reinterpret_cast<const u32x4*>(bPtr0);
  u32x4 pb1 = *reinterpret_cast<const u32x4*>(bPtr1);

  for (int kb = 0; kb < K; kb += 32) {
    *reinterpret_cast<u32x4*>(sA0) = pa0;
    *reinterpret_cast<u32x4*>(sA1) = pa1;
    *reinterpret_cast<u32x4*>(sB0) = pb0;
    *reinterpret_cast<u32x4*>(sB1) = pb1;
    __syncthreads();
    if (kb + 32 < K) {   // prefetch next k-block; overlaps with WMMA below
      pa0 = *reinterpret_cast<const u32x4*>(aPtr0 + kb + 32);
      pa1 = *reinterpret_cast<const u32x4*>(aPtr1 + kb + 32);
      pb0 = *reinterpret_cast<const u32x4*>(bPtr0 + kb + 32);
      pb1 = *reinterpret_cast<const u32x4*>(bPtr1 + kb + 32);
    }
    v16bf bfr[2];
#pragma unroll
    for (int nt = 0; nt < 2; ++nt)
      bfr[nt] = load_fragB(&lB[(wn * 32 + nt * 16 + l16) * LSTR], half);
#pragma unroll
    for (int mt = 0; mt < 4; ++mt) {
      v16bf af = load_fragA(&lA[(wm * 64 + mt * 16 + l16) * LSTR], half);
      acc[mt][0] = wmma_bf16(af, bfr[0], acc[mt][0]);
      acc[mt][1] = wmma_bf16(af, bfr[1], acc[mt][1]);
    }
    __syncthreads();
  }

#pragma unroll
  for (int mt = 0; mt < 4; ++mt)
#pragma unroll
    for (int nt = 0; nt < 2; ++nt) {
      int gn = n0 + wn * 32 + nt * 16 + l16;
      int gmbase = m0 + wm * 64 + mt * 16 + half * 8;   // 8 consecutive rows
      if (MODE == 1) {
#pragma unroll
        for (int r = 0; r < 8; ++r)
          outF[(size_t)(gmbase + r) * N + gn] = acc[mt][nt][r];
      } else {
        int bb = gmbase >> 11, q = gmbase & (L_ - 1);   // tile never crosses batch
        int sect = gn >> 10, nl = gn & (D_ - 1);
        int hh = nl >> 6, hd = nl & (HD_ - 1);
        size_t bhq = (size_t)(bb * H_ + hh);
        if (sect == 0) {
#pragma unroll
          for (int r = 0; r < 8; ++r)
            q_out[(bhq * L_ + q + r) * HD_ + hd] = (bf16_t)(acc[mt][nt][r] * QSCALE);
        } else if (sect == 1) {
#pragma unroll
          for (int r = 0; r < 8; ++r)
            k_out[(bhq * L_ + q + r) * HD_ + hd] = (bf16_t)acc[mt][nt][r];
        } else {  // V^T: q contiguous -> packed 16-B store
          v8bf pk;
#pragma unroll
          for (int r = 0; r < 8; ++r) pk[r] = (bf16_t)acc[mt][nt][r];
          *reinterpret_cast<v8bf*>(&vt_out[(bhq * HD_ + hd) * L_ + q]) = pk;
        }
      }
    }
}

// ---------- flash attention, transposed orientation ----------
// S^T = K Q^T  (lane col = query q, VGPR rows = key j)  -> softmax stats per lane
// ctx^T = V^T P^T (lane col = q, VGPR rows = d)         -> rescale/normalize per lane
// One wave handles 32 query rows (2 q-tiles) sharing K / V^T fragments.
__global__ __launch_bounds__(128) void attn_kernel(
    const bf16_t* __restrict__ Q,   // [BH, L, 64], pre-scaled by log2e/8
    const bf16_t* __restrict__ Kb,  // [BH, L, 64]
    const bf16_t* __restrict__ Vt,  // [BH, 64, L]
    bf16_t* __restrict__ Ctx) {     // [B*L, 1024], col = h*64 + d
  __shared__ bf16_t pbuf_all[4][2][16 * LSTR];
  const int lane = threadIdx.x & 31;
  const int wave = threadIdx.x >> 5;
  const int l16  = lane & 15;
  const int half = lane >> 4;

  int tile = blockIdx.x * 4 + wave;   // 4096 tiles of 32 query rows
  int bh = tile >> 6;                 // 64 tiles per (b,h)
  int q0 = (tile & 63) * 32;
  int b = bh >> 4, h = bh & 15;
  const bf16_t* Qp = Q  + (size_t)bh * L_ * HD_;
  const bf16_t* Kp = Kb + (size_t)bh * L_ * HD_;
  const bf16_t* Vp = Vt + (size_t)bh * HD_ * L_;

  // Q^T B-fragments: qf[qt][db]; lane col q = q0 + qt*16 + l16, k = d interleaved
  v16bf qf[2][2];
#pragma unroll
  for (int qt = 0; qt < 2; ++qt) {
    const bf16_t* qrow = Qp + (size_t)(q0 + qt * 16 + l16) * HD_;
    qf[qt][0] = load_fragB(qrow, half);
    qf[qt][1] = load_fragB(qrow + 32, half);
  }

  v8f acc[2][4];
#pragma unroll
  for (int qt = 0; qt < 2; ++qt)
#pragma unroll
    for (int dt = 0; dt < 4; ++dt) acc[qt][dt] = (v8f){0,0,0,0,0,0,0,0};
  float mq[2] = { -__builtin_inff(), -__builtin_inff() };
  float lq[2] = { 0.0f, 0.0f };

  const int jb_end = q0 + 32;          // keys j <= q0+31 needed
  for (int jb = 0; jb < jb_end; jb += 32) {
    if (jb + 32 < jb_end) {            // CDNA5 global_prefetch of next K/V block
      __builtin_prefetch(Kp + (size_t)(jb + 32 + lane) * HD_, 0, 3);
      __builtin_prefetch(Vp + (size_t)lane * L_ + jb + 32, 0, 3);
      __builtin_prefetch(Vp + (size_t)(lane + 32) * L_ + jb + 32, 0, 3);
    }
    // K A-fragments (rows j, k = d interleaved), shared by both q-tiles
    v16bf ka[2][2];
#pragma unroll
    for (int jt = 0; jt < 2; ++jt) {
      const bf16_t* krow = Kp + (size_t)(jb + jt * 16 + l16) * HD_;
      ka[jt][0] = load_fragA(krow, half);
      ka[jt][1] = load_fragA(krow + 32, half);
    }
    // V^T A-fragments (rows d, k = j interleaved), shared
    v16bf va[4];
#pragma unroll
    for (int dt = 0; dt < 4; ++dt)
      va[dt] = load_fragA(Vp + (size_t)(dt * 16 + l16) * L_ + jb, half);

#pragma unroll
    for (int qt = 0; qt < 2; ++qt) {
      const int qv = q0 + qt * 16 + l16;   // this lane's query index
      v8f s[2];
#pragma unroll
      for (int jt = 0; jt < 2; ++jt) {
        v8f c = (v8f){0,0,0,0,0,0,0,0};
        c = wmma_bf16(ka[jt][0], qf[qt][0], c);
        c = wmma_bf16(ka[jt][1], qf[qt][1], c);
        s[jt] = c;
      }
      // causal mask: element (j = jb + jt*16 + 8*half + r, q = qv)
      if (jb + 31 > q0 + qt * 16) {
#pragma unroll
        for (int jt = 0; jt < 2; ++jt) {
          int jbase = jb + jt * 16 + 8 * half;
#pragma unroll
          for (int r = 0; r < 8; ++r)
            if (jbase + r > qv) s[jt][r] = -__builtin_inff();
        }
      }
      // online softmax: 15 in-lane max + ONE cross-half shuffle
      float mx = s[0][0];
#pragma unroll
      for (int jt = 0; jt < 2; ++jt)
#pragma unroll
        for (int r = 0; r < 8; ++r) mx = fmaxf(mx, s[jt][r]);
      mx = fmaxf(mx, __shfl_xor(mx, 16, 32));
      float mnew = fmaxf(mq[qt], mx);
      float sc = exp2f(mq[qt] - mnew);
      float rs = 0.0f;
#pragma unroll
      for (int jt = 0; jt < 2; ++jt)
#pragma unroll
        for (int r = 0; r < 8; ++r) {
          float p = exp2f(s[jt][r] - mnew);
          s[jt][r] = p;
          rs += p;
        }
      rs += __shfl_xor(rs, 16, 32);
      lq[qt] = lq[qt] * sc + rs;
      mq[qt] = mnew;
#pragma unroll
      for (int dt = 0; dt < 4; ++dt)
#pragma unroll
        for (int r = 0; r < 8; ++r) acc[qt][dt][r] *= sc;

      // stage P^T into LDS as [q][j] rows: one packed 16-B store per jt-tile
      bf16_t* pb = pbuf_all[wave][qt];
#pragma unroll
      for (int jt = 0; jt < 2; ++jt) {
        v8bf pk;
#pragma unroll
        for (int r = 0; r < 8; ++r) pk[r] = (bf16_t)s[jt][r];
        *reinterpret_cast<v8bf*>(&pb[l16 * LSTR + jt * 16 + 8 * half]) = pk;
      }
    }
    asm volatile("s_wait_dscnt 0" ::: "memory");   // wave-local LDS RAW fence
#pragma unroll
    for (int qt = 0; qt < 2; ++qt) {
      v16bf pfrag = load_fragB(&pbuf_all[wave][qt][l16 * LSTR], half);  // P^T as B operand
#pragma unroll
      for (int dt = 0; dt < 4; ++dt)
        acc[qt][dt] = wmma_bf16(va[dt], pfrag, acc[qt][dt]);
    }
  }

  // normalize + store ctx^T: lane col q, rows d contiguous -> packed 16-B stores
#pragma unroll
  for (int qt = 0; qt < 2; ++qt) {
    float inv = 1.0f / lq[qt];
    size_t rowbase = ((size_t)b * L_ + q0 + qt * 16 + l16) * D_ + h * HD_;
#pragma unroll
    for (int dt = 0; dt < 4; ++dt) {
      v8bf pk;
#pragma unroll
      for (int r = 0; r < 8; ++r) pk[r] = (bf16_t)(acc[qt][dt][r] * inv);
      *reinterpret_cast<v8bf*>(&Ctx[rowbase + dt * 16 + 8 * half]) = pk;
    }
  }
}

// ---------- launcher ----------
extern "C" void kernel_launch(void* const* d_in, const int* in_sizes, int n_in,
                              void* d_out, int out_size, void* d_ws, size_t ws_size,
                              hipStream_t stream) {
  const float* x    = (const float*)d_in[0];   // [B,L,D]
  const float* wqkv = (const float*)d_in[1];   // [3D,D]
  const float* wout = (const float*)d_in[2];   // [D,D]
  float* out = (float*)d_out;

  char* ws = (char*)d_ws;
  size_t o = 0;
  bf16_t* Xbf    = (bf16_t*)(ws + o); o += (size_t)M_ROWS * D_ * 2;
  bf16_t* Wqkvbf = (bf16_t*)(ws + o); o += (size_t)3 * D_ * D_ * 2;
  bf16_t* Woutbf = (bf16_t*)(ws + o); o += (size_t)D_ * D_ * 2;
  bf16_t* Qbf    = (bf16_t*)(ws + o); o += (size_t)BH_ * L_ * HD_ * 2;
  bf16_t* Kbf    = (bf16_t*)(ws + o); o += (size_t)BH_ * L_ * HD_ * 2;
  bf16_t* Vtbf   = (bf16_t*)(ws + o); o += (size_t)BH_ * HD_ * L_ * 2;
  bf16_t* Ctxbf  = (bf16_t*)(ws + o); o += (size_t)M_ROWS * D_ * 2;

  // stage 0: convert inputs to bf16
  cvt_kernel<<<M_ROWS * D_ / 4 / 256, 256, 0, stream>>>(x, Xbf, M_ROWS * D_ / 4);
  cvt_kernel<<<3 * D_ * D_ / 4 / 256, 256, 0, stream>>>(wqkv, Wqkvbf, 3 * D_ * D_ / 4);
  cvt_kernel<<<D_ * D_ / 4 / 256, 256, 0, stream>>>(wout, Woutbf, D_ * D_ / 4);

  // stage 1: fused QKV projection -> Q (scaled) / K / V^T
  gemm_kernel<0><<<dim3(3 * D_ / 128, M_ROWS / 128), 256, 0, stream>>>(
      Xbf, Wqkvbf, nullptr, Qbf, Kbf, Vtbf, M_ROWS, 3 * D_, D_);

  // stage 2: causal flash attention (transposed softmax orientation)
  attn_kernel<<<(BH_ * (L_ / 32)) / 4, 128, 0, stream>>>(Qbf, Kbf, Vtbf, Ctxbf);

  // stage 3: output projection (fp32 out)
  gemm_kernel<1><<<dim3(D_ / 128, M_ROWS / 128), 256, 0, stream>>>(
      Ctxbf, Woutbf, out, nullptr, nullptr, nullptr, M_ROWS, D_, D_);
}
